// LinearShiftQ_32487132627344
// MI455X (gfx1250) — matrix operand order repaired and checked by reading, not compiled, not run on hardware
//
#include <hip/hip_runtime.h>
#include <hip/hip_bf16.h>

typedef __attribute__((ext_vector_type(16))) __bf16 v16bf;
typedef __attribute__((ext_vector_type(8)))  float  v8f;

#define BM 128
#define BN 128
#define BK 64
#define LDP (BK + 8)   // LDS row pitch in bf16 elements (16B padding kills bank conflicts)

// --- elementwise quantizers --------------------------------------------------

// DeepShift round_to_fixed: floor to 2^-16 grid, clamp to signed 16.16 range
__device__ __forceinline__ float round_fixed(float x) {
  float r = floorf(x * 65536.0f) * (1.0f / 65536.0f);
  return fminf(fmaxf(r, -32768.0f), 32767.0f);
}

// f32 -> bf16 round-to-nearest-even (bit trick)
__device__ __forceinline__ unsigned short f32_to_bf16_rne(float f) {
  unsigned u = __float_as_uint(f);
  u += 0x7FFFu + ((u >> 16) & 1u);
  return (unsigned short)(u >> 16);
}

// sign(w) * 2^round(log2|w|): keep sign+exponent, bump exponent if mantissa>=sqrt(2)'s.
// Result has zero mantissa -> top 16 bits are the *exact* bf16 encoding.
__device__ __forceinline__ unsigned short pow2_quant_bf16(float w) {
  unsigned u    = __float_as_uint(w);
  unsigned mant = u & 0x007FFFFFu;
  unsigned se   = u & 0xFF800000u;           // sign | exponent
  if (mant >= 0x3504F4u) se += 0x00800000u;  // round(log2(m)) == 1 when m >= sqrt(2)
  return (unsigned short)(se >> 16);
}

// 16-element load+convert helper: 4x float4 -> 16 packed bf16 (two uint4)
struct Pk { uint4 u[2]; };

template <bool WQ>
__device__ __forceinline__ Pk load16(const float* __restrict__ g) {
  union { unsigned short s[16]; Pk p; } t;
#pragma unroll
  for (int q = 0; q < 4; ++q) {
    float4 f = ((const float4*)g)[q];
    if (WQ) {
      t.s[q * 4 + 0] = pow2_quant_bf16(f.x);
      t.s[q * 4 + 1] = pow2_quant_bf16(f.y);
      t.s[q * 4 + 2] = pow2_quant_bf16(f.z);
      t.s[q * 4 + 3] = pow2_quant_bf16(f.w);
    } else {
      t.s[q * 4 + 0] = f32_to_bf16_rne(round_fixed(f.x));
      t.s[q * 4 + 1] = f32_to_bf16_rne(round_fixed(f.y));
      t.s[q * 4 + 2] = f32_to_bf16_rne(round_fixed(f.z));
      t.s[q * 4 + 3] = f32_to_bf16_rne(round_fixed(f.w));
    }
  }
  return t.p;
}

// --- GEMM kernel -------------------------------------------------------------
// out[M,N] = bf16(round_fixed(X[M,K])) @ bf16(pow2(W[N,K]))^T + round_fixed(bias[N])
__global__ __launch_bounds__(256) void deepshift_gemm(
    const float* __restrict__ X, const float* __restrict__ W,
    const float* __restrict__ Bias, float* __restrict__ Out,
    int M, int N, int K) {
  __shared__ alignas(16) unsigned short As[BM * LDP];
  __shared__ alignas(16) unsigned short Bs[BN * LDP];

  const int tid  = threadIdx.x;
  const int lane = tid & 31;
  const int wave = tid >> 5;      // 8 waves
  const int wm   = wave & 1;      // 2 waves along M
  const int wn   = wave >> 1;     // 4 waves along N
  const int half = lane >> 4;     // 0: lanes 0-15, 1: lanes 16-31
  const int l16  = lane & 15;

  const int m0 = blockIdx.y * BM;
  const int n0 = blockIdx.x * BN;

  // tile-load chunk coordinates for this thread (2 chunks of 16 per tile)
  const int r0 = tid >> 2, c0 = (tid & 3) << 4;          // chunk 0: rows 0..63
  const int r1 = (tid + 256) >> 2, c1 = ((tid) & 3) << 4; // chunk 1: rows 64..127

  v8f acc[4][2];
#pragma unroll
  for (int i = 0; i < 4; ++i)
#pragma unroll
    for (int j = 0; j < 2; ++j) {
      v8f z = {0.f, 0.f, 0.f, 0.f, 0.f, 0.f, 0.f, 0.f};
      acc[i][j] = z;
    }

  for (int k0 = 0; k0 < K; k0 += BK) {
    // stage global loads + convert in registers (overlaps with previous MMA phase)
    Pk a0 = load16<false>(X + (size_t)(m0 + r0) * K + k0 + c0);
    Pk a1 = load16<false>(X + (size_t)(m0 + r1) * K + k0 + c1);
    Pk b0 = load16<true >(W + (size_t)(n0 + r0) * K + k0 + c0);
    Pk b1 = load16<true >(W + (size_t)(n0 + r1) * K + k0 + c1);

    // prefetch next K-tile (global_prefetch_b8)
    if (k0 + BK < K) {
      __builtin_prefetch(X + (size_t)(m0 + r0) * K + k0 + BK + c0, 0, 0);
      __builtin_prefetch(X + (size_t)(m0 + r1) * K + k0 + BK + c1, 0, 0);
      __builtin_prefetch(W + (size_t)(n0 + r0) * K + k0 + BK + c0, 0, 0);
      __builtin_prefetch(W + (size_t)(n0 + r1) * K + k0 + BK + c1, 0, 0);
    }

    __syncthreads();  // previous compute phase done -> safe to overwrite LDS
    *(uint4*)&As[r0 * LDP + c0]     = a0.u[0];
    *(uint4*)&As[r0 * LDP + c0 + 8] = a0.u[1];
    *(uint4*)&As[r1 * LDP + c1]     = a1.u[0];
    *(uint4*)&As[r1 * LDP + c1 + 8] = a1.u[1];
    *(uint4*)&Bs[r0 * LDP + c0]     = b0.u[0];
    *(uint4*)&Bs[r0 * LDP + c0 + 8] = b0.u[1];
    *(uint4*)&Bs[r1 * LDP + c1]     = b1.u[0];
    *(uint4*)&Bs[r1 * LDP + c1 + 8] = b1.u[1];
    __syncthreads();

    // two k-steps of 32 per tile
#pragma unroll
    for (int kk = 0; kk < BK; kk += 32) {
      v16bf a[4], b[2];
#pragma unroll
      for (int i = 0; i < 4; ++i) {
        // A frag (16x32 bf16): lane half0 holds K[0..7]+K[16..23], half1 K[8..15]+K[24..31]
        const int row = wm * 64 + i * 16 + l16;
        const int kb  = kk + half * 8;
        union { uint4 u[2]; v16bf v; } t;
        t.u[0] = *(const uint4*)&As[row * LDP + kb];
        t.u[1] = *(const uint4*)&As[row * LDP + kb + 16];
        a[i] = t.v;
      }
#pragma unroll
      for (int j = 0; j < 2; ++j) {
        // B frag (32x16 bf16): lane = column, half0 holds K[0..7]+K[16..23], half1 K[8..15]+K[24..31]
        const int col = wn * 32 + j * 16 + l16;
        const int kb  = kk + half * 8;
        union { uint4 u[2]; v16bf v; } t;
        t.u[0] = *(const uint4*)&Bs[col * LDP + kb];
        t.u[1] = *(const uint4*)&Bs[col * LDP + kb + 16];
        b[j] = t.v;
      }
#pragma unroll
      for (int i = 0; i < 4; ++i)
#pragma unroll
        for (int j = 0; j < 2; ++j)
          acc[i][j] = __builtin_amdgcn_wmma_f32_16x16x32_bf16(
              false, a[i], false, b[j], (short)0, acc[i][j], false, false);
    }
    __syncthreads();
  }

  // epilogue: C/D layout = lane(0-15)=N col; VGPR v -> M = v + 8*half
#pragma unroll
  for (int j = 0; j < 2; ++j) {
    const int n  = n0 + wn * 32 + j * 16 + l16;
    const float bq = round_fixed(Bias[n]);
#pragma unroll
    for (int i = 0; i < 4; ++i) {
      const int mbase = m0 + wm * 64 + i * 16 + half * 8;
#pragma unroll
      for (int v = 0; v < 8; ++v)
        Out[(size_t)(mbase + v) * N + n] = acc[i][j][v] + bq;
    }
  }
}

extern "C" void kernel_launch(void* const* d_in, const int* in_sizes, int n_in,
                              void* d_out, int out_size, void* d_ws, size_t ws_size,
                              hipStream_t stream) {
  const float* X    = (const float*)d_in[0];
  const float* Wt   = (const float*)d_in[1];
  const float* Bias = (const float*)d_in[2];
  float* Out        = (float*)d_out;

  const int OUT_F = in_sizes[2];              // 4096
  const int IN_F  = in_sizes[1] / OUT_F;      // 4096
  const int M     = in_sizes[0] / IN_F;       // 8192

  dim3 grid(OUT_F / BN, M / BM);              // 32 x 64 workgroups
  deepshift_gemm<<<grid, 256, 0, stream>>>(X, Wt, Bias, Out, M, OUT_F, IN_F);
}
